// TurnTakingDetector_40003325395705
// MI455X (gfx1250) — compile-verified, matrix operand order checked by reading, not run on hardware
//
#include <hip/hip_runtime.h>
#include <stdint.h>

// ---------------------------------------------------------------------------
// Bidirectional LSTM (B=32, T=1024, D=1024, H=512) + FC(1) + sigmoid
// Phase 0a: pack Wx/Wh fp32 -> bf16 in WMMA B-operand tile layout
// Phase 0b: convert x fp32 -> bf16 (memory-bound, removes VALU cvt from GEMM)
// Phase 1 : xg = x @ Wx + b, stored in WMMA C-fragment layout (32B/lane),
//           non-temporal stores (streaming, keep Wh hot in L2)
// Phase 2 : sequential recurrence, 4 workgroups (2 dirs x 2 batch halves),
//           h in LDS, c in VGPRs, WMMA bf16 per step, fused FC reduction,
//           NT fragment C-init + next-step prefetch
// Phase 3 : out[b][t] = sigmoid(logitF + logitB + b_fc)
// ---------------------------------------------------------------------------

#define T_LEN 1024
#define B_SZ  32
#define D_IN  1024
#define H_SZ  512
#define G4    2048              // 4*H
#define NT    128               // G4/16 n-tiles
#define KT_X  32                // D_IN/32 k-tiles
#define KT_H  16                // H_SZ/32 k-tiles

typedef __attribute__((ext_vector_type(16))) __bf16         v16bf;
typedef __attribute__((ext_vector_type(16))) unsigned short v16u;
typedef __attribute__((ext_vector_type(8)))  unsigned short v8u;
typedef __attribute__((ext_vector_type(8)))  float          v8f;
typedef __attribute__((ext_vector_type(4)))  float          v4f;

static constexpr size_t XG_ELEMS = (size_t)T_LEN * B_SZ * G4;   // fragment layout
static constexpr size_t X_ELEMS  = (size_t)B_SZ * T_LEN * D_IN;

__device__ __forceinline__ unsigned short f2bf(float f) {
  union { float f; unsigned u; } cv; cv.f = f;
  unsigned u = cv.u;
  u += 0x7FFFu + ((u >> 16) & 1u);   // round-to-nearest-even
  return (unsigned short)(u >> 16);
}

__device__ __forceinline__ float sigmoidf_(float x) {
  return __builtin_amdgcn_rcpf(1.0f + __expf(-x));
}
__device__ __forceinline__ float tanhf_(float x) {
  // tanh(x) = 2*sigmoid(2x) - 1 ; exp overflow/underflow saturates correctly
  return 2.0f * __builtin_amdgcn_rcpf(1.0f + __expf(-2.0f * x)) - 1.0f;
}

// ---------------------------------------------------------------------------
// Phase 0a: pack W[K][G4] (fp32 row-major) into bf16 WMMA B-operand tiles.
// Tile (nt,kt) = 32x16 bf16 = 1KB; lane l owns 32 contiguous bytes holding
// its 16 v16bf elements (K-pairs per VGPR, halves split at K=8).
// ---------------------------------------------------------------------------
__global__ void pack_weights_kernel(const float* __restrict__ W,
                                    unsigned short* __restrict__ dst,
                                    int KT) {
  int idx = blockIdx.x * blockDim.x + threadIdx.x;
  int total = NT * KT * 32;
  if (idx >= total) return;
  int lane = idx & 31;
  int kt   = (idx >> 5) % KT;
  int nt   = idx / (KT * 32);
  int n     = nt * 16 + (lane & 15);
  int kbase = kt * 32 + ((lane < 16) ? 0 : 8);
  unsigned short* out = dst + (size_t)idx * 16;
#pragma unroll
  for (int e = 0; e < 16; ++e) {
    int k = kbase + ((e < 8) ? e : (e + 8));
    out[e] = f2bf(W[(size_t)k * G4 + n]);
  }
}

// ---------------------------------------------------------------------------
// Phase 0b: x fp32 -> bf16, 8 elements per thread (memory bound).
// ---------------------------------------------------------------------------
__global__ void cvt_x_kernel(const float* __restrict__ x,
                             unsigned short* __restrict__ xbf) {
  size_t i = ((size_t)blockIdx.x * blockDim.x + threadIdx.x) * 8;
  if (i >= X_ELEMS) return;
  v4f a0 = *(const v4f*)(x + i);
  v4f a1 = *(const v4f*)(x + i + 4);
  v8u o;
#pragma unroll
  for (int e = 0; e < 4; ++e) {
    o[e]     = f2bf(a0[e]);
    o[e + 4] = f2bf(a1[e]);
  }
  *(v8u*)(xbf + i) = o;
}

// ---------------------------------------------------------------------------
// Phase 1: xg tile (mt,nt) = x-rows @ Wx + bias, stored as C fragments:
//   xg[(((size_t)mt*NT + nt)*32 + lane)*8 .. +8)   (32B per lane, coalesced)
// m-tile rows r = mt*16 + i are (t = mt/2, b = (mt&1)*16 + i)  -> aligns
// exactly with phase-2 per-(t, batch-half) C tiles.
// ---------------------------------------------------------------------------
__global__ __launch_bounds__(256)
void xg_gemm_kernel(const unsigned short* __restrict__ xbf,
                    const unsigned short* __restrict__ WxP,
                    const float* __restrict__ bias,
                    float* __restrict__ xg) {
  const int wave = threadIdx.x >> 5;
  const int lane = threadIdx.x & 31;
  const int ng = blockIdx.x & 15;
  const int mt = blockIdx.x >> 4;
  const int nt = ng * 8 + wave;
  const int m0 = mt * 16;
  const int r  = m0 + (lane & 15);
  const int t  = r >> 5;
  const int b  = r & 31;
  const unsigned short* __restrict__ xrow =
      xbf + ((size_t)b * T_LEN + t) * D_IN;
  const int c0 = (lane < 16) ? 0 : 8;
  const unsigned short* __restrict__ bp =
      WxP + (size_t)nt * KT_X * 512 + (size_t)lane * 16;

  v8f acc = {0.f, 0.f, 0.f, 0.f, 0.f, 0.f, 0.f, 0.f};
#pragma unroll 4
  for (int kt = 0; kt < KT_X; ++kt) {
    const int k0 = kt * 32;
    const v8u lo = *(const v8u*)(xrow + k0 + c0);
    const v8u hi = *(const v8u*)(xrow + k0 + c0 + 16);
    const v16u au = __builtin_shufflevector(
        lo, hi, 0, 1, 2, 3, 4, 5, 6, 7, 8, 9, 10, 11, 12, 13, 14, 15);
    const v16bf av = __builtin_bit_cast(v16bf, au);
    const v16bf bv = *(const v16bf*)(bp + (size_t)kt * 512);
    acc = __builtin_amdgcn_wmma_f32_16x16x32_bf16(
        false, av, false, bv, (short)0, acc, false, false);
  }
  const float bn = bias[nt * 16 + (lane & 15)];
#pragma unroll
  for (int v = 0; v < 8; ++v) acc[v] += bn;
  // fragment store: 32B contiguous per lane, non-temporal (streamed once)
  __builtin_nontemporal_store(
      acc, (v8f*)(xg + (((size_t)mt * NT + nt) * 32 + lane) * 8));
}

// ---------------------------------------------------------------------------
// Phase 2: LSTM recurrence. Grid (2 batch-halves, 2 directions), 512 thr.
// Wave w owns hidden units j in [32w, 32w+32): its 8 n-tiles are the
// i/f/g/o columns for exactly those units -> gate fusion is lane-local.
// ---------------------------------------------------------------------------
__global__ __launch_bounds__(512)
void lstm_rec_kernel(const float* __restrict__ xgF, const float* __restrict__ xgB,
                     const unsigned short* __restrict__ WhPf,
                     const unsigned short* __restrict__ WhPb,
                     const float* __restrict__ wfc,
                     float* __restrict__ logitF, float* __restrict__ logitB) {
  const int rev = blockIdx.y;
  const float* __restrict__ xg = rev ? xgB : xgF;
  const unsigned short* __restrict__ WhP = rev ? WhPb : WhPf;
  float* __restrict__ logit = rev ? logitB : logitF;

  const int half = blockIdx.x;          // batch half
  const int b0 = half * 16;
  const int wave = threadIdx.x >> 5;
  const int lane = threadIdx.x & 31;
  const int ci = lane & 15;
  const int c0 = (lane < 16) ? 0 : 8;
  const int rowadd = (lane < 16) ? 0 : 8;

  __shared__ unsigned short h_lds[16][H_SZ + 8];   // +8 pad vs bank conflicts
  __shared__ float logit_acc[16];

  for (int i = threadIdx.x; i < 16 * (H_SZ + 8); i += 512)
    (&h_lds[0][0])[i] = 0;
  if (threadIdx.x < 16) logit_acc[threadIdx.x] = 0.f;
  __syncthreads();

  const float wfc0 = wfc[rev * H_SZ + wave * 32 + ci];
  const float wfc1 = wfc[rev * H_SZ + wave * 32 + 16 + ci];

  v8f cst0 = {0.f, 0.f, 0.f, 0.f, 0.f, 0.f, 0.f, 0.f};
  v8f cst1 = {0.f, 0.f, 0.f, 0.f, 0.f, 0.f, 0.f, 0.f};

  for (int s = 0; s < T_LEN; ++s) {
    const int t = rev ? (T_LEN - 1 - s) : s;
    const int mt = t * 2 + half;

    // C init = xg fragments (bias folded in phase 1): 8 x v8f NT loads
    v8f acc[8];
#pragma unroll
    for (int g = 0; g < 4; ++g) {
#pragma unroll
      for (int u = 0; u < 2; ++u) {
        const int nt = g * 32 + 2 * wave + u;
        acc[g * 2 + u] = __builtin_nontemporal_load(
            (const v8f*)(xg + (((size_t)mt * NT + nt) * 32 + lane) * 8));
      }
    }

    // prefetch next timestep's xg fragments (hidden behind the GEMM below)
    const int tn = rev ? (t - 1) : (t + 1);
    if (tn >= 0 && tn < T_LEN) {
      const int mtn = tn * 2 + half;
#pragma unroll
      for (int g = 0; g < 4; ++g) {
#pragma unroll
        for (int u = 0; u < 2; ++u) {
          const int nt = g * 32 + 2 * wave + u;
          __builtin_prefetch(xg + (((size_t)mtn * NT + nt) * 32 + lane) * 8,
                             0, 1);
        }
      }
    }

    // g += h_{t-1} @ Wh : 16 k-tiles x 8 n-tiles of WMMA
    // (no explicit Wh prefetch: Wh is L2-resident; compiler pipelines loads)
#pragma unroll 2
    for (int kt = 0; kt < KT_H; ++kt) {
      const int k0 = kt * 32;
      const v8u lo = *(const v8u*)&h_lds[ci][k0 + c0];
      const v8u hi = *(const v8u*)&h_lds[ci][k0 + 16 + c0];
      const v16u au = __builtin_shufflevector(
          lo, hi, 0, 1, 2, 3, 4, 5, 6, 7, 8, 9, 10, 11, 12, 13, 14, 15);
      const v16bf av = __builtin_bit_cast(v16bf, au);
#pragma unroll
      for (int g = 0; g < 4; ++g) {
#pragma unroll
        for (int u = 0; u < 2; ++u) {
          const int nt = g * 32 + 2 * wave + u;
          const v16bf bv = *(const v16bf*)(
              WhP + ((size_t)(nt * KT_H + kt) * 32 + lane) * 16);
          acc[g * 2 + u] = __builtin_amdgcn_wmma_f32_16x16x32_bf16(
              false, av, false, bv, (short)0, acc[g * 2 + u], false, false);
        }
      }
    }

    __syncthreads();   // all waves done reading h_lds

    float lsum[8];
#pragma unroll
    for (int v = 0; v < 8; ++v) lsum[v] = 0.f;

#pragma unroll
    for (int u = 0; u < 2; ++u) {
      const v8f iv = acc[0 + u];
      const v8f fv = acc[2 + u];
      const v8f gv = acc[4 + u];
      const v8f ov = acc[6 + u];
      const int j = wave * 32 + u * 16 + ci;
      const float wf = u ? wfc1 : wfc0;
      v8f* cp = u ? &cst1 : &cst0;
#pragma unroll
      for (int v = 0; v < 8; ++v) {
        const float ii = sigmoidf_(iv[v]);
        const float ff = sigmoidf_(fv[v]);
        const float gg = tanhf_(gv[v]);
        const float oo = sigmoidf_(ov[v]);
        const float c  = ff * (*cp)[v] + ii * gg;
        (*cp)[v] = c;
        const float h = oo * tanhf_(c);
        h_lds[v + rowadd][j] = f2bf(h);       // publish h_t for next step
        lsum[v] += h * wf;                    // fused FC partial
      }
    }

    // reduce FC partials over the 16 lanes of each half-wave (same batch rows)
#pragma unroll
    for (int off = 1; off < 16; off <<= 1) {
#pragma unroll
      for (int v = 0; v < 8; ++v)
        lsum[v] += __shfl_xor(lsum[v], off, 32);
    }
    if (ci == 0) {
#pragma unroll
      for (int v = 0; v < 8; ++v)
        atomicAdd(&logit_acc[v + rowadd], lsum[v]);   // ds_add_f32
    }

    __syncthreads();   // h writes + atomics done

    if (threadIdx.x < 16) {
      logit[(size_t)t * B_SZ + b0 + threadIdx.x] = logit_acc[threadIdx.x];
      logit_acc[threadIdx.x] = 0.f;           // next-step atomics gated by next sync
    }
  }
}

// ---------------------------------------------------------------------------
// Phase 3: out[b][t] = sigmoid(logitF[t][b] + logitB[t][b] + b_fc)
// ---------------------------------------------------------------------------
__global__ void final_kernel(const float* __restrict__ logitF,
                             const float* __restrict__ logitB,
                             const float* __restrict__ bfc,
                             float* __restrict__ out) {
  int idx = blockIdx.x * blockDim.x + threadIdx.x;
  if (idx >= B_SZ * T_LEN) return;
  int b = idx / T_LEN;
  int t = idx - b * T_LEN;
  float z = logitF[t * B_SZ + b] + logitB[t * B_SZ + b] + bfc[0];
  out[idx] = 1.0f / (1.0f + __expf(-z));
}

// ---------------------------------------------------------------------------
extern "C" void kernel_launch(void* const* d_in, const int* in_sizes, int n_in,
                              void* d_out, int out_size, void* d_ws, size_t ws_size,
                              hipStream_t stream) {
  const float* x    = (const float*)d_in[0];
  const float* Wx_f = (const float*)d_in[1];
  const float* Wh_f = (const float*)d_in[2];
  const float* b_f  = (const float*)d_in[3];
  const float* Wx_b = (const float*)d_in[4];
  const float* Wh_b = (const float*)d_in[5];
  const float* b_b  = (const float*)d_in[6];
  const float* W_fc = (const float*)d_in[7];
  const float* b_fc = (const float*)d_in[8];

  char* ws = (char*)d_ws;
  size_t off = 0;
  float* xgF = (float*)(ws + off); off += XG_ELEMS * sizeof(float);
  float* xgB = (float*)(ws + off); off += XG_ELEMS * sizeof(float);
  unsigned short* xbf  = (unsigned short*)(ws + off); off += X_ELEMS * 2;
  unsigned short* WxPf = (unsigned short*)(ws + off); off += (size_t)NT * KT_X * 512 * 2;
  unsigned short* WxPb = (unsigned short*)(ws + off); off += (size_t)NT * KT_X * 512 * 2;
  unsigned short* WhPf = (unsigned short*)(ws + off); off += (size_t)NT * KT_H * 512 * 2;
  unsigned short* WhPb = (unsigned short*)(ws + off); off += (size_t)NT * KT_H * 512 * 2;
  float* logitF = (float*)(ws + off); off += (size_t)T_LEN * B_SZ * sizeof(float);
  float* logitB = (float*)(ws + off); off += (size_t)T_LEN * B_SZ * sizeof(float);

  const int packX = NT * KT_X * 32;  // 131072 threads
  const int packH = NT * KT_H * 32;  //  65536 threads
  pack_weights_kernel<<<(packX + 255) / 256, 256, 0, stream>>>(Wx_f, WxPf, KT_X);
  pack_weights_kernel<<<(packX + 255) / 256, 256, 0, stream>>>(Wx_b, WxPb, KT_X);
  pack_weights_kernel<<<(packH + 255) / 256, 256, 0, stream>>>(Wh_f, WhPf, KT_H);
  pack_weights_kernel<<<(packH + 255) / 256, 256, 0, stream>>>(Wh_b, WhPb, KT_H);

  const size_t cvtThreads = X_ELEMS / 8;     // 8 elems/thread
  cvt_x_kernel<<<(unsigned)((cvtThreads + 255) / 256), 256, 0, stream>>>(x, xbf);

  xg_gemm_kernel<<<2048 * 16, 256, 0, stream>>>(xbf, WxPf, b_f, xgF);
  xg_gemm_kernel<<<2048 * 16, 256, 0, stream>>>(xbf, WxPb, b_b, xgB);

  lstm_rec_kernel<<<dim3(2, 2), 512, 0, stream>>>(xgF, xgB, WhPf, WhPb,
                                                  W_fc, logitF, logitB);

  final_kernel<<<(B_SZ * T_LEN + 255) / 256, 256, 0, stream>>>(
      logitF, logitB, b_fc, (float*)d_out);
}
